// FNO2d_36249523978358
// MI455X (gfx1250) — matrix operand order repaired and checked by reading, not compiled, not run on hardware
//
#include <hip/hip_runtime.h>
#include <math.h>

// ---------------- problem constants ----------------
#define BATCH   8
#define CH      64
#define HH      256
#define WW      256
#define NLAYER  4
#define MODES   16     // MX = MY = 16

// ---------------- workspace layout (bytes) ----------------
// h buffers: [B][H][W][64] bf16, channel-last (GEMM fuel lives narrow in memory)
#define OFF_H0   0ULL
#define OFF_H1   (OFF_H0 + 67108864ULL)                  // 64 MB each
// R: fwd-W result, bf16, B-style [b][c][ky16][512] (k<256: Rr over h, k>=256: Ri)
#define OFF_R    (OFF_H1 + 67108864ULL)                  // 8 MB
// Fm: fwd-H result, bf16, per mode B-style [mode256][bb16][128] (k<64: Fr, k>=64: Fi)
#define OFF_FM   (OFF_R + 8388608ULL)                    // 1 MB
// Ghat: mode-mix result, bf16, B-style [b][co][ky16][32] (k<16: Or over kx, k>=16: Oi)
#define OFF_GH   (OFF_FM + 1048576ULL)                   // 512 KB
// Ginv: inv-H result, bf16, B-style [b][h][c][32] (k<16: Gr over ky, k>=16: Gi)
#define OFF_GI   (OFF_GH + 524288ULL)                    // 8 MB
// bf16 DFT basis tables
#define OFF_TWF  (OFF_GI + 8388608ULL)                   // [32][256]  fwd-W basis (B-style)
#define OFF_THFR (OFF_TWF + 16384ULL)                    // [16][512]  fwd-H basis for Fr (A-style)
#define OFF_THFI (OFF_THFR + 16384ULL)                   // [16][512]  fwd-H basis for Fi (A-style)
#define OFF_TIR  (OFF_THFI + 16384ULL)                   // [256][32]  inv-H basis for Gr (A-style)
#define OFF_TII  (OFF_TIR + 16384ULL)                    // [256][32]  inv-H basis for Gi (A-style)
#define OFF_TWI  (OFF_TII + 16384ULL)                    // [256][32]  inv-W basis (A-style, alpha folded)
// bf16 weight mirrors
#define OFF_PWB  (OFF_TWI + 16384ULL)                    // [NL][64][64] pointwise W (B-style [co][ci])
#define OFF_O1B  (OFF_PWB + 32768ULL)                    // [32][64]     head W1    (B-style [o][ci])

// ---------------- WMMA types & helpers ----------------
typedef __attribute__((ext_vector_type(16))) __bf16         v16bf;
typedef __attribute__((ext_vector_type(8)))  float          v8f;
typedef __attribute__((ext_vector_type(16))) unsigned short us16;
typedef __attribute__((ext_vector_type(8)))  unsigned short us8;
typedef __attribute__((ext_vector_type(4)))  unsigned int   tdm_u32x4;
typedef __attribute__((ext_vector_type(4)))  int            tdm_i32x4;
typedef __attribute__((ext_vector_type(8)))  int            tdm_i32x8;

__device__ __forceinline__ unsigned short f2bfbits(float f) {
    unsigned int u = __builtin_bit_cast(unsigned int, f);
    unsigned int r = (u + 0x7fffu + ((u >> 16) & 1u)) >> 16;   // RNE
    return (unsigned short)r;
}

__device__ __forceinline__ float gelu_exact(float x) {
    return 0.5f * x * (1.0f + erff(x * 0.70710678118654752440f));
}

// A fragment (bf16 16x32): storage row-major [m][k], k contiguous.
// lane<16 : M=lane,    K = {0..15}
// lane>=16: M=lane-16, K = {16..31}
__device__ __forceinline__ v16bf ldfragA(const unsigned short* base, int stride,
                                         int m0, int k0, int lane) {
    int r = lane & 15, half = lane >> 4;
    const unsigned short* p = base + (long long)(m0 + r) * stride + k0 + half * 16;
    us8 lo = *(const us8*)p;
    us8 hi = *(const us8*)(p + 8);
    us16 v = __builtin_shufflevector(lo, hi, 0,1,2,3,4,5,6,7,8,9,10,11,12,13,14,15);
    return __builtin_bit_cast(v16bf, v);
}

// B fragment (bf16 32x16): storage [n][k] ("B-style"), k contiguous.
__device__ __forceinline__ v16bf ldfragB(const unsigned short* base, int stride,
                                         int n0, int k0, int lane) {
    int r = lane & 15, half = lane >> 4;
    const unsigned short* p = base + (long long)(n0 + r) * stride + k0 + half * 16;
    return __builtin_bit_cast(v16bf, *(const us16*)p);
}

__device__ __forceinline__ v8f wmma_bf(v16bf a, v16bf b, v8f c) {
    return __builtin_amdgcn_wmma_f32_16x16x32_bf16(false, a, false, b,
                                                   (short)0, c, false, false);
}

// -------- Tensor Data Mover: DMA a contiguous run of bf16 elems into LDS --------
// Builds a D# per CDNA5 ISA 8.3/8.4: count=1, type=2("image"), data_size=2B,
// 1-D tile (tile_dim1=0 unused), tensor_dim0 = tile_dim0 = nelem.
// ldsByteOff is relative to the block's LDS allocation base (single __shared__
// array per kernel => array base is offset 0).
#if __has_builtin(__builtin_amdgcn_tensor_load_to_lds)
#define HAVE_TDM 1
#if __clang_major__ >= 23
#warning "CDNA5 probe: TDM active, 6-arg builtin (clang>=23 / amdgpu-toolchain)"
#else
#warning "CDNA5 probe: TDM active, 5-arg builtin (clang-22 / ROCm 7.2)"
#endif
__device__ __forceinline__ void tdm_load_bf16_1d(const unsigned short* gptr,
                                                 unsigned ldsByteOff, unsigned nelem) {
    unsigned long long ga = (unsigned long long)(size_t)gptr;
    tdm_u32x4 g0;
    g0.x = 1u;                                              // count=1, user mode
    g0.y = ldsByteOff;                                      // lds_addr
    g0.z = (unsigned)(ga & 0xffffffffull);                  // global_addr[31:0]
    g0.w = (unsigned)((ga >> 32) & 0x01ffffffull)           // global_addr[56:32]
           | 0x80000000u;                                   // type=2
    tdm_i32x8 g1;
    g1[0] = 0x00010000;                                     // data_size=1 (2 bytes)
    g1[1] = (int)((nelem & 0xffffu) << 16);                 // tensor_dim0[15:0]
    g1[2] = (int)((nelem >> 16) | (1u << 16));              // tensor_dim0[31:16], tensor_dim1=1
    g1[3] = (int)((nelem & 0xffffu) << 16);                 // tile_dim0 = nelem
    g1[4] = 0;                                              // tile_dim1=0 (unused), tile_dim2=0
    g1[5] = (int)nelem;                                     // tensor_dim0_stride[31:0]
    g1[6] = (int)((nelem & 0xffffu) << 16);                 // t0s[47:32]=0, t1s[15:0]
    g1[7] = (int)(nelem >> 16);                             // t1s[47:16]
    tdm_i32x4 z4 = {0, 0, 0, 0};
#if __clang_major__ >= 23
    tdm_i32x8 z8 = {0, 0, 0, 0, 0, 0, 0, 0};
    __builtin_amdgcn_tensor_load_to_lds(g0, g1, z4, z4, z8, 0);
#else
    __builtin_amdgcn_tensor_load_to_lds(g0, g1, z4, z4, 0);
#endif
}
#else
#define HAVE_TDM 0
#warning "CDNA5 probe: tensor_load_to_lds builtin NOT available; manual LDS copy fallback in use"
#endif

// ---------------- K0: build DFT basis tables + bf16 weight mirrors ----------------
__global__ void k0_init(char* ws, const float* pw_w, const float* o1_w) {
    unsigned short* TWF  = (unsigned short*)(ws + OFF_TWF);
    unsigned short* THFR = (unsigned short*)(ws + OFF_THFR);
    unsigned short* THFI = (unsigned short*)(ws + OFF_THFI);
    unsigned short* TIR  = (unsigned short*)(ws + OFF_TIR);
    unsigned short* TII  = (unsigned short*)(ws + OFF_TII);
    unsigned short* TWI  = (unsigned short*)(ws + OFF_TWI);
    unsigned short* FM   = (unsigned short*)(ws + OFF_FM);
    unsigned short* PWB  = (unsigned short*)(ws + OFF_PWB);
    unsigned short* O1B  = (unsigned short*)(ws + OFF_O1B);

    const float S  = 1.0f / 16.0f;   // sqrt split of the 1/256 ortho factor per stage
    const float TP = 6.28318530717958647692f;
    int tid = blockIdx.x * blockDim.x + threadIdx.x;
    int nth = gridDim.x * blockDim.x;

    for (int i = tid; i < 32 * 256; i += nth) {              // TW_f [32][256]
        int n = i >> 8, w = i & 255;
        float th = TP * (float)((n & 15) * w) / 256.0f;
        TWF[i] = f2bfbits((n < 16) ? cosf(th) * S : -sinf(th) * S);
    }
    for (int i = tid; i < 16 * 512; i += nth) {              // TH_f_r / TH_f_i [16][512]
        int m = i >> 9, k = i & 511, hh = k & 255;
        float th = TP * (float)(m * hh) / 256.0f;
        THFR[i] = f2bfbits((k < 256) ? cosf(th) * S :  sinf(th) * S);
        THFI[i] = f2bfbits((k < 256) ? -sinf(th) * S : cosf(th) * S);
    }
    for (int i = tid; i < 256 * 32; i += nth) {              // TH_i_r / TH_i_i [256][32]
        int hh = i >> 5, k = i & 31, kx = k & 15;
        float th = TP * (float)(kx * hh) / 256.0f;
        TIR[i] = f2bfbits((k < 16) ? cosf(th) * S : -sinf(th) * S);
        TII[i] = f2bfbits((k < 16) ? sinf(th) * S :  cosf(th) * S);
    }
    for (int i = tid; i < 256 * 32; i += nth) {              // TW_i [256][32]
        int w = i >> 5, k = i & 31, ky = k & 15;
        float alpha = (ky == 0) ? 1.0f : 2.0f;
        float th = TP * (float)(ky * w) / 256.0f;
        TWI[i] = f2bfbits((k < 16) ? alpha * cosf(th) * S : -alpha * sinf(th) * S);
    }
    for (int i = tid; i < 256 * 16 * 128; i += nth) FM[i] = 0;   // zero pad rows bb=8..15
    for (int i = tid; i < NLAYER * 64 * 64; i += nth) PWB[i] = f2bfbits(pw_w[i]);
    for (int i = tid; i < 32 * 64; i += nth)          O1B[i] = f2bfbits(o1_w[i]);
}

// ---------------- K1: lift conv1x1 2->64 (memory bound), writes bf16 h ----------
__global__ void k1_lift(const float* __restrict__ x, const float* __restrict__ in_w,
                        const float* __restrict__ in_b, unsigned short* __restrict__ h) {
    long long idx = (long long)blockIdx.x * blockDim.x + threadIdx.x;
    long long N = (long long)BATCH * HH * WW * CH;
    long long stride = (long long)gridDim.x * blockDim.x;
    for (; idx < N; idx += stride) {
        int c = (int)(idx & 63);
        long long p = idx >> 6;
        int b = (int)(p >> 16);
        int rem = (int)(p & 65535);
        float x0 = x[((long long)b * 2 + 0) * 65536 + rem];
        float x1 = x[((long long)b * 2 + 1) * 65536 + rem];
        h[idx] = f2bfbits(in_w[c * 2 + 0] * x0 + in_w[c * 2 + 1] * x1 + in_b[c]);
    }
}

// ---------------- K2: forward DFT along W (K=256 GEMM, 16 modes x {r,i}) ---------
__global__ void k2_fwdW(const unsigned short* __restrict__ h, char* ws) {
    __shared__ unsigned short ldsA[64 * 256];        // A-style [c][w], 32 KB
    int bh = blockIdx.x;                             // b*256 + hrow
    int b = bh >> 8, hrow = bh & 255;
    const unsigned short* hp = h + (long long)bh * 256 * 64;
    for (int i = threadIdx.x; i < 256 * 64; i += blockDim.x) {
        int c = i & 63, w = i >> 6;
        ldsA[c * 256 + w] = hp[i];                   // transpose to [c][w], bf16 passthrough
    }
    __syncthreads();

    const unsigned short* TWF = (const unsigned short*)(ws + OFF_TWF);
    unsigned short* R = (unsigned short*)(ws + OFF_R);
    int lane = threadIdx.x & 31, wv = threadIdx.x >> 5;
    int ct = wv & 3, nt = wv >> 2;                   // 4 c-tiles x 2 n-tiles = 8 waves

    v8f acc = {};
    for (int ks = 0; ks < 8; ++ks) {
        __builtin_prefetch(TWF + (nt * 16) * 256 + ((ks + 1) & 7) * 32, 0, 1);
        v16bf a = ldfragA(ldsA, 256, ct * 16, ks * 32, lane);
        v16bf bb = ldfragB(TWF, 256, nt * 16, ks * 32, lane);
        acc = wmma_bf(a, bb, acc);
    }
    int r = lane & 15, half = lane >> 4;
    for (int v = 0; v < 8; ++v) {
        int c = ct * 16 + v + half * 8;
        int ky = r;
        int col = (nt == 0) ? hrow : (256 + hrow);   // Rr | Ri
        R[(((long long)b * 64 + c) * 16 + ky) * 512 + col] = f2bfbits(acc[v]);
    }
}

// ---------------- K3: forward DFT along H (K=512 GEMM -> 16x16 mode corner) ------
__global__ void k3_fwdH(char* ws) {
    int bc = blockIdx.x;                             // b*64 + c
    const unsigned short* R = (const unsigned short*)(ws + OFF_R) + (long long)bc * 16 * 512;
    unsigned short* FM = (unsigned short*)(ws + OFF_FM);
    int lane = threadIdx.x & 31, wv = threadIdx.x >> 5;   // 2 waves: Fr | Fi
    const unsigned short* A =
        (const unsigned short*)(ws + (wv == 0 ? OFF_THFR : OFF_THFI));

    v8f acc = {};
    for (int ks = 0; ks < 16; ++ks) {
        __builtin_prefetch(R + ((ks + 1) & 15) * 32, 0, 1);
        v16bf a = ldfragA(A, 512, 0, ks * 32, lane);
        v16bf bb = ldfragB(R, 512, 0, ks * 32, lane);
        acc = wmma_bf(a, bb, acc);
    }
    int r = lane & 15, half = lane >> 4;
    int b = bc >> 6, c = bc & 63;
    for (int v = 0; v < 8; ++v) {
        int kx = v + half * 8, ky = r;
        int mode = kx * 16 + ky;
        FM[(long long)mode * 2048 + b * 128 + (wv == 0 ? c : 64 + c)] = f2bfbits(acc[v]);
    }
}

// ---------------- K4: per-mode complex channel mix (K=128 folded complex GEMM) ---
__global__ void k4_modemix(char* ws, const float* __restrict__ spec_wr,
                           const float* __restrict__ spec_wi, int layer) {
    __shared__ unsigned short ldsAB[2 * 64 * 128];   // A1 | A2, 32 KB
    unsigned short* A1 = ldsAB;
    unsigned short* A2 = ldsAB + 64 * 128;
    int mode = blockIdx.x;
    const float* wr = spec_wr + (long long)layer * 64 * 64 * 256;
    const float* wi = spec_wi + (long long)layer * 64 * 64 * 256;
    for (int i = threadIdx.x; i < 64 * 64; i += blockDim.x) {
        int co = i >> 6, ci = i & 63;
        float vr = wr[((long long)ci * 64 + co) * 256 + mode];
        float vi = wi[((long long)ci * 64 + co) * 256 + mode];
        A1[co * 128 + ci]      = f2bfbits(vr);
        A1[co * 128 + 64 + ci] = f2bfbits(-vi);
        A2[co * 128 + ci]      = f2bfbits(vi);
        A2[co * 128 + 64 + ci] = f2bfbits(vr);
    }
    __syncthreads();

    const unsigned short* FM = (const unsigned short*)(ws + OFF_FM) + (long long)mode * 2048;
    unsigned short* GH = (unsigned short*)(ws + OFF_GH);
    int lane = threadIdx.x & 31, wv = threadIdx.x >> 5;
    int ct = wv & 3, which = wv >> 2;                // 4 co-tiles x {Or,Oi}
    const unsigned short* A = which ? A2 : A1;

    v8f acc = {};
    for (int ks = 0; ks < 4; ++ks) {
        v16bf a = ldfragA(A, 128, ct * 16, ks * 32, lane);
        v16bf bb = ldfragB(FM, 128, 0, ks * 32, lane);
        acc = wmma_bf(a, bb, acc);
    }
    int r = lane & 15, half = lane >> 4;
    int kx = mode >> 4, ky = mode & 15;
    if (r < 8) {                                     // only bb=0..7 valid
        for (int v = 0; v < 8; ++v) {
            int co = ct * 16 + v + half * 8;
            GH[(((long long)r * 64 + co) * 16 + ky) * 32 + (which ? 16 + kx : kx)] =
                f2bfbits(acc[v]);
        }
    }
}

// ---------------- K5: inverse DFT along H (K=32 GEMM) ----------------------------
__global__ void k5_invH(char* ws) {
    int bco = blockIdx.x;                            // b*64 + co
    int b = bco >> 6, co = bco & 63;
    const unsigned short* GH  = (const unsigned short*)(ws + OFF_GH) + (long long)bco * 16 * 32;
    const unsigned short* TIR = (const unsigned short*)(ws + OFF_TIR);
    const unsigned short* TII = (const unsigned short*)(ws + OFF_TII);
    unsigned short* GI = (unsigned short*)(ws + OFF_GI);
    int lane = threadIdx.x & 31, wv = threadIdx.x >> 5;

    v16bf bfrag = ldfragB(GH, 32, 0, 0, lane);       // shared across all tiles
    for (int t = 0; t < 4; ++t) {
        int task = wv * 4 + t;                       // 16 h-tiles x {Gr,Gi} = 32
        int mt = task >> 1, which = task & 1;
        const unsigned short* A = which ? TII : TIR;
        v8f acc = {};
        v16bf a = ldfragA(A, 32, mt * 16, 0, lane);
        acc = wmma_bf(a, bfrag, acc);
        int r = lane & 15, half = lane >> 4;
        for (int v = 0; v < 8; ++v) {
            int hrow = mt * 16 + v + half * 8;
            int ky = r;
            GI[(((long long)(b * 256 + hrow)) * 64 + co) * 32 + (which ? 16 + ky : ky)] =
                f2bfbits(acc[v]);
        }
    }
}

// -------- K6: fused inverse-W DFT + pointwise conv + bias + exact GELU -----------
// Activation row staged LDS-direct via the Tensor Data Mover (no VGPR staging).
__global__ void k6_spatial(const unsigned short* __restrict__ hin,
                           unsigned short* __restrict__ hout,
                           char* ws, const float* __restrict__ pw_b, int layer) {
    __shared__ unsigned short hrowL[256 * 64];       // A-style [w][ci], 32 KB, LDS offset 0
    int bh = blockIdx.x;                             // b*256 + hrow
    const unsigned short* hp = hin + (long long)bh * 256 * 64;
#if HAVE_TDM
    if ((threadIdx.x >> 5) == 0) {                   // one TDM op per block
        tdm_load_bf16_1d(hp, 0u, 256u * 64u);
        __builtin_amdgcn_s_wait_tensorcnt(0);
    }
#else
    for (int i = threadIdx.x; i < 256 * 64; i += blockDim.x) hrowL[i] = hp[i];
#endif
    __syncthreads();

    const unsigned short* TWI = (const unsigned short*)(ws + OFF_TWI);
    const unsigned short* GI  = (const unsigned short*)(ws + OFF_GI) + (long long)bh * 64 * 32;
    const unsigned short* PWB = (const unsigned short*)(ws + OFF_PWB) + (long long)layer * 64 * 64;
    const float* pb = pw_b + layer * 64;
    int lane = threadIdx.x & 31, wv = threadIdx.x >> 5;

    for (int j = 0; j < 8; ++j) {
        int tile = wv * 8 + j;                       // 16 w-tiles x 4 c-tiles = 64
        int wt = tile >> 2, ct = tile & 3;
        v8f acc = {};
        {   // spectral part x1: D[w][c] = TW_i[w][k32] * Ginv[c][k32]
            v16bf a  = ldfragA(TWI, 32, wt * 16, 0, lane);
            v16bf bb = ldfragB(GI, 32, ct * 16, 0, lane);
            acc = wmma_bf(a, bb, acc);
        }
        for (int ks = 0; ks < 2; ++ks) {             // pointwise x2: K=64
            __builtin_prefetch(PWB + (ct * 16) * 64 + ((ks + 1) & 1) * 32, 0, 1);
            v16bf a  = ldfragA(hrowL, 64, wt * 16, ks * 32, lane);
            v16bf bb = ldfragB(PWB, 64, ct * 16, ks * 32, lane);
            acc = wmma_bf(a, bb, acc);
        }
        int r = lane & 15, half = lane >> 4;
        for (int v = 0; v < 8; ++v) {
            int w = wt * 16 + v + half * 8;
            int c = ct * 16 + r;
            float val = acc[v] + pb[c];
            hout[(long long)bh * 256 * 64 + (long long)w * 64 + c] =
                f2bfbits(gelu_exact(val));
        }
    }
}

// ---------------- K7: head (64->32 WMMA + GELU, then 32->1 VALU dot) -------------
__global__ void k7_head(const unsigned short* __restrict__ hin,
                        const float* __restrict__ o1_b,
                        const float* __restrict__ o2_w, const float* __restrict__ o2_b,
                        float* __restrict__ out, char* ws) {
    __shared__ __align__(16) unsigned char ldsraw[32768 + 32768];  // single LDS alloc
    unsigned short* hrowL = (unsigned short*)ldsraw;               // offset 0
    float* tbuf = (float*)(ldsraw + 32768);
    int bh = blockIdx.x;
    const unsigned short* hp = hin + (long long)bh * 256 * 64;
#if HAVE_TDM
    if ((threadIdx.x >> 5) == 0) {
        tdm_load_bf16_1d(hp, 0u, 256u * 64u);
        __builtin_amdgcn_s_wait_tensorcnt(0);
    }
#else
    for (int i = threadIdx.x; i < 256 * 64; i += blockDim.x) hrowL[i] = hp[i];
#endif
    __syncthreads();

    const unsigned short* O1B = (const unsigned short*)(ws + OFF_O1B);
    int lane = threadIdx.x & 31, wv = threadIdx.x >> 5;
    for (int t = 0; t < 4; ++t) {
        int task = wv * 4 + t;                       // 16 w-tiles x 2 j-tiles = 32
        int wt = task >> 1, nt = task & 1;
        v8f acc = {};
        for (int ks = 0; ks < 2; ++ks) {
            v16bf a  = ldfragA(hrowL, 64, wt * 16, ks * 32, lane);
            v16bf bb = ldfragB(O1B, 64, nt * 16, ks * 32, lane);
            acc = wmma_bf(a, bb, acc);
        }
        int r = lane & 15, half = lane >> 4;
        for (int v = 0; v < 8; ++v) {
            int w = wt * 16 + v + half * 8;
            int jj = nt * 16 + r;
            tbuf[w * 32 + jj] = gelu_exact(acc[v] + o1_b[jj]);
        }
    }
    __syncthreads();
    int w = threadIdx.x;                             // one lane per w
    float s = o2_b[0];
    for (int j = 0; j < 32; ++j) s += o2_w[j] * tbuf[w * 32 + j];
    out[(long long)bh * 256 + w] = s;
}

// ---------------- host-side launch ----------------
extern "C" void kernel_launch(void* const* d_in, const int* in_sizes, int n_in,
                              void* d_out, int out_size, void* d_ws, size_t ws_size,
                              hipStream_t stream) {
    const float* x       = (const float*)d_in[0];
    const float* in_w    = (const float*)d_in[1];
    const float* in_b    = (const float*)d_in[2];
    const float* spec_wr = (const float*)d_in[3];
    const float* spec_wi = (const float*)d_in[4];
    const float* pw_w    = (const float*)d_in[5];
    const float* pw_b    = (const float*)d_in[6];
    const float* o1_w    = (const float*)d_in[7];
    const float* o1_b    = (const float*)d_in[8];
    const float* o2_w    = (const float*)d_in[9];
    const float* o2_b    = (const float*)d_in[10];
    float* out = (float*)d_out;
    char* ws   = (char*)d_ws;

    k0_init<<<1024, 256, 0, stream>>>(ws, pw_w, o1_w);

    unsigned short* h0 = (unsigned short*)(ws + OFF_H0);
    unsigned short* h1 = (unsigned short*)(ws + OFF_H1);
    k1_lift<<<16384, 256, 0, stream>>>(x, in_w, in_b, h0);

    unsigned short* hcur = h0;
    unsigned short* hnxt = h1;
    for (int l = 0; l < NLAYER; ++l) {
        k2_fwdW<<<BATCH * HH, 256, 0, stream>>>(hcur, ws);
        k3_fwdH<<<BATCH * CH, 64, 0, stream>>>(ws);
        k4_modemix<<<MODES * MODES, 256, 0, stream>>>(ws, spec_wr, spec_wi, l);
        k5_invH<<<BATCH * CH, 256, 0, stream>>>(ws);
        k6_spatial<<<BATCH * HH, 256, 0, stream>>>(hcur, hnxt, ws, pw_b, l);
        unsigned short* t = hcur; hcur = hnxt; hnxt = t;
    }
    k7_head<<<BATCH * HH, 256, 0, stream>>>(hcur, o1_b, o2_w, o2_b, out, ws);
}
